// DDRec_76201309766071
// MI455X (gfx1250) — matrix operand order repaired and checked by reading, not compiled, not run on hardware
//
#include <hip/hip_runtime.h>

#define NU_   32768
#define NI_   8192
#define NN_   (NU_ + NI_)
#define DIM_  64
#define OUTD_ 192

typedef _Float16 f16_t;
typedef __attribute__((ext_vector_type(16))) _Float16 v16h;
typedef __attribute__((ext_vector_type(8)))  float    v8f;

// ------------------------------ helpers ------------------------------------
__device__ __forceinline__ float wave_sum32(float s) {
#pragma unroll
  for (int o = 16; o > 0; o >>= 1) s += __shfl_xor(s, o, 32);
  return s;
}

__device__ __forceinline__ void atomic_add_f(float* p, float v) {
  unsafeAtomicAdd(p, v);   // lowers to global_atomic_add_f32
}

__global__ void fill_f32(float* __restrict__ p, int n, float v) {
  int i = blockIdx.x * blockDim.x + threadIdx.x;
  if (i < n) p[i] = v;
}

// ego/accum init: concat(user_emb, feat) along rows
__global__ void init_ego(const float* __restrict__ ue, const float* __restrict__ feat,
                         float* __restrict__ ego, float* __restrict__ accum) {
  int i = blockIdx.x * blockDim.x + threadIdx.x;
  if (i >= NN_ * DIM_) return;
  int r = i >> 6;
  float v = (r < NU_) ? ue[i] : feat[i - NU_ * DIM_];
  ego[i] = v;
  accum[i] = v;
}

// --------------------- WMMA GEMM: C[Mx64] = X[MxK] @ W[64xK]^T (+bias) -----
// One wave computes a 16-row x 64-col strip. K must be a multiple of 32.
// A layout (16-bit A 16x32): lane l<16 half=0 holds K 0..15 ascending,
// half=1 holds K 16..31 (matches B). B layout (K x 16, same contiguous k order as A):
// lanes 0-15 hold K 0..15 ascending, lanes 16-31 hold K 16..31.
// D layout: vgpr i, lane: row = i + 8*(lane/16), col = lane%16.
__global__ void __launch_bounds__(256)
gemm_xwt_f16(const float* __restrict__ X, const float* __restrict__ W,
             const float* __restrict__ bias, float* __restrict__ C,
             int M, int K) {
  const int wave = threadIdx.x >> 5;
  const int lane = threadIdx.x & 31;
  const int l    = lane & 15;
  const int half = lane >> 4;
  const int m0   = blockIdx.x * 128 + wave * 16;
  if (m0 >= M) return;  // wave-uniform: EXEC stays all-ones for WMMA

  v8f acc[4];
#pragma unroll
  for (int t = 0; t < 4; ++t)
#pragma unroll
    for (int i = 0; i < 8; ++i) acc[t][i] = 0.0f;

  const float* xrow = X + (size_t)(m0 + l) * K;

  for (int kk = 0; kk < K; kk += 32) {
    const int kb0 = kk + half * 16;
    const int kb1 = kk + half * 16 + 8;
    v16h a;
#pragma unroll
    for (int j = 0; j < 8; ++j) {
      a[j]     = (f16_t)xrow[kb0 + j];
      a[8 + j] = (f16_t)xrow[kb1 + j];
    }
#pragma unroll
    for (int t = 0; t < 4; ++t) {
      const float* wrow = W + (size_t)(t * 16 + l) * K + kk + half * 16;
      v16h b;
#pragma unroll
      for (int j = 0; j < 16; ++j) b[j] = (f16_t)wrow[j];
      acc[t] = __builtin_amdgcn_wmma_f32_16x16x32_f16(
          false, a, false, b, (short)0, acc[t], false, false);
    }
  }

#pragma unroll
  for (int t = 0; t < 4; ++t) {
#pragma unroll
    for (int i = 0; i < 8; ++i) {
      int r = m0 + half * 8 + i;
      int c = t * 16 + l;
      float v = acc[t][i];
      if (bias) v += bias[c];
      C[(size_t)r * DIM_ + c] = v;
    }
  }
}

// ---------------- KNN SpMM (rows are np.repeat-structured) ------------------
// One wave per item row; lane handles dims {lane, lane+32}. mm graph passes
// part2_off = nnz_img (second contiguous segment of 10 entries per row).
__global__ void spmm_knn(const int* __restrict__ cols, const float* __restrict__ vals,
                         const float* __restrict__ Hin, float* __restrict__ Hout,
                         int k, int part2_off) {
  int w = (blockIdx.x * blockDim.x + threadIdx.x) >> 5;
  int lane = threadIdx.x & 31;
  if (w >= NI_) return;
  float s0 = 0.f, s1 = 0.f;
  int base = w * k;
  for (int j = 0; j < k; ++j) {
    int e = base + j;
    float v = vals[e];
    size_t c = (size_t)cols[e] * DIM_;
    s0 += v * Hin[c + lane];
    s1 += v * Hin[c + 32 + lane];
  }
  if (part2_off >= 0) {
    for (int j = 0; j < k; ++j) {
      int e = part2_off + base + j;
      float v = vals[e];
      size_t c = (size_t)cols[e] * DIM_;
      s0 += v * Hin[c + lane];
      s1 += v * Hin[c + 32 + lane];
    }
  }
  Hout[(size_t)w * DIM_ + lane]      = s0;
  Hout[(size_t)w * DIM_ + 32 + lane] = s1;
}

__global__ void l2norm_rows(float* __restrict__ H, int rows) {
  int w = (blockIdx.x * blockDim.x + threadIdx.x) >> 5;
  int lane = threadIdx.x & 31;
  if (w >= rows) return;
  size_t b = (size_t)w * DIM_;
  float a0 = H[b + lane], a1 = H[b + 32 + lane];
  float s = wave_sum32(a0 * a0 + a1 * a1);
  float scale = 1.f / fmaxf(sqrtf(s), 1e-12f);
  H[b + lane]      = a0 * scale;
  H[b + 32 + lane] = a1 * scale;
}

// --------- per-edge mask (filtered dot >= 0) + degree accumulation ---------
__global__ void edge_mask_deg(const int* __restrict__ eu, const int* __restrict__ ev,
                              const float* __restrict__ ego, float* __restrict__ mask,
                              float* __restrict__ deg, int E, int filtered) {
  int e = (blockIdx.x * blockDim.x + threadIdx.x) >> 5;
  int lane = threadIdx.x & 31;
  if (e >= E) return;
  int u = eu[e], v = ev[e];
  float m = 1.0f;
  if (filtered) {
    size_t ub = (size_t)u * DIM_, vb = (size_t)v * DIM_;
    float s = ego[ub + lane] * ego[vb + lane] +
              ego[ub + 32 + lane] * ego[vb + 32 + lane];
    s = wave_sum32(s);
    m = (s >= 0.0f) ? 1.0f : 0.0f;
  }
  if (lane == 0) {
    mask[e] = m;
    if (m != 0.0f) {
      atomic_add_f(&deg[v], m);
      atomic_add_f(&deg[u], m);
    }
  }
}

__global__ void make_dinv(const float* __restrict__ deg, float* __restrict__ dinv, int n) {
  int i = blockIdx.x * blockDim.x + threadIdx.x;
  if (i < n) dinv[i] = rsqrtf(deg[i] + 1.0f);
}

// ----- symmetric-normalized aggregation over both edge directions ----------
__global__ void aggregate(const int* __restrict__ eu, const int* __restrict__ ev,
                          const float* __restrict__ mask, const float* __restrict__ dinv,
                          const float* __restrict__ hlin, float* __restrict__ agg, int E) {
  int e = (blockIdx.x * blockDim.x + threadIdx.x) >> 5;
  int lane = threadIdx.x & 31;
  if (e >= E) return;
  float m = mask[e];
  if (m == 0.0f) return;  // wave-uniform
  int u = eu[e], v = ev[e];
  float c = m * dinv[u] * dinv[v];
  size_t ub = (size_t)u * DIM_, vb = (size_t)v * DIM_;
  float hu0 = hlin[ub + lane], hu1 = hlin[ub + 32 + lane];
  float hv0 = hlin[vb + lane], hv1 = hlin[vb + 32 + lane];
  atomic_add_f(&agg[vb + lane],      hu0 * c);
  atomic_add_f(&agg[vb + 32 + lane], hu1 * c);
  atomic_add_f(&agg[ub + lane],      hv0 * c);
  atomic_add_f(&agg[ub + 32 + lane], hv1 * c);
}

// ego_new = agg + hlin*dinv^2 + b ; accum += ego_new
__global__ void combine(const float* __restrict__ agg, const float* __restrict__ hlin,
                        const float* __restrict__ dinv, const float* __restrict__ bias,
                        float* __restrict__ ego, float* __restrict__ accum) {
  int i = blockIdx.x * blockDim.x + threadIdx.x;
  if (i >= NN_ * DIM_) return;
  int r = i >> 6, c = i & 63;
  float di = dinv[r];
  float v = agg[i] + hlin[i] * di * di + bias[c];
  ego[i] = v;
  accum[i] += v;
}

// out[:, colbase:colbase+64] = accum/3 (+ mmfeat for item rows)
__global__ void write_out(const float* __restrict__ accum, const float* __restrict__ mmfeat,
                          float* __restrict__ out, int colbase) {
  int i = blockIdx.x * blockDim.x + threadIdx.x;
  if (i >= NN_ * DIM_) return;
  int r = i >> 6, c = i & 63;
  float v = accum[i] * (1.0f / 3.0f);
  if (r >= NU_) v += mmfeat[(size_t)(r - NU_) * DIM_ + c];
  out[(size_t)r * OUTD_ + colbase + c] = v;
}

// ------------------------------ host side ----------------------------------
static void run_propagate(const float* user_emb, const float* feat,
                          const float* conv_W, const float* conv_b,
                          const int* eu, const int* ev, int E, int filtered,
                          const float* mmfeat, int colbase,
                          float* ego, float* hlin, float* agg, float* accum,
                          float* deg, float* dinv, float* mask,
                          float* out, hipStream_t stream) {
  const int TPB = 256;
  const int nElem = NN_ * DIM_;
  const int elemBlocks = (nElem + TPB - 1) / TPB;
  const int edgeBlocks = (E + 7) / 8;          // 8 waves per block

  init_ego<<<elemBlocks, TPB, 0, stream>>>(user_emb, feat, ego, accum);

  for (int l = 0; l < 2; ++l) {
    fill_f32<<<(NN_ + TPB - 1) / TPB, TPB, 0, stream>>>(deg, NN_, 0.0f);
    edge_mask_deg<<<edgeBlocks, TPB, 0, stream>>>(eu, ev, ego, mask, deg, E, filtered);
    make_dinv<<<(NN_ + TPB - 1) / TPB, TPB, 0, stream>>>(deg, dinv, NN_);

    gemm_xwt_f16<<<(NN_ + 127) / 128, TPB, 0, stream>>>(
        ego, conv_W + (size_t)l * DIM_ * DIM_, nullptr, hlin, NN_, DIM_);

    fill_f32<<<elemBlocks, TPB, 0, stream>>>(agg, nElem, 0.0f);
    aggregate<<<edgeBlocks, TPB, 0, stream>>>(eu, ev, mask, dinv, hlin, agg, E);
    combine<<<elemBlocks, TPB, 0, stream>>>(agg, hlin, dinv,
                                            conv_b + (size_t)l * DIM_, ego, accum);
  }
  write_out<<<elemBlocks, TPB, 0, stream>>>(accum, mmfeat, out, colbase);
}

extern "C" void kernel_launch(void* const* d_in, const int* in_sizes, int n_in,
                              void* d_out, int out_size, void* d_ws, size_t ws_size,
                              hipStream_t stream) {
  const float* user_emb = (const float*)d_in[0];
  const float* item_emb = (const float*)d_in[1];
  const float* v_feat   = (const float*)d_in[2];
  const float* t_feat   = (const float*)d_in[3];
  const float* Wi       = (const float*)d_in[4];
  const float* bi       = (const float*)d_in[5];
  const float* Wt       = (const float*)d_in[6];
  const float* bt       = (const float*)d_in[7];
  const float* conv_W   = (const float*)d_in[8];
  const float* conv_b   = (const float*)d_in[9];
  const int*   edge     = (const int*)d_in[10];
  const int*   img_idx  = (const int*)d_in[11];
  const float* img_vals = (const float*)d_in[12];
  const int*   txt_idx  = (const int*)d_in[13];
  const float* txt_vals = (const float*)d_in[14];
  const int*   mm_idx   = (const int*)d_in[15];
  const float* mm_vals  = (const float*)d_in[16];
  float* out = (float*)d_out;

  const int E       = in_sizes[10] / 2;
  const int nnz_img = in_sizes[12];          // NI*10
  const int Kv      = in_sizes[2] / NI_;     // 1024
  const int Kt      = in_sizes[3] / NI_;     // 384
  const int knn_k   = nnz_img / NI_;         // 10

  const int* eu = edge;
  const int* ev = edge + E;

  // workspace carve-out (fp32), 256B-aligned chunks
  float* base = (float*)d_ws;
  size_t off = 0;
  auto alloc = [&](size_t n) {
    float* p = base + off;
    off += (n + 63) & ~(size_t)63;
    return p;
  };
  float* v_emb = alloc((size_t)NI_ * DIM_);
  float* t_emb = alloc((size_t)NI_ * DIM_);
  float* hmm   = alloc((size_t)NI_ * DIM_);
  float* h1    = alloc((size_t)NI_ * DIM_);
  float* h2    = alloc((size_t)NI_ * DIM_);
  float* ego   = alloc((size_t)NN_ * DIM_);
  float* hlin  = alloc((size_t)NN_ * DIM_);
  float* agg   = alloc((size_t)NN_ * DIM_);
  float* accum = alloc((size_t)NN_ * DIM_);
  float* deg   = alloc(NN_);
  float* dinv  = alloc(NN_);
  float* mask  = alloc((size_t)E);
  (void)ws_size; (void)n_in; (void)out_size;

  const int TPB = 256;

  // ---- dense multimodal projections via WMMA ----
  gemm_xwt_f16<<<(NI_ + 127) / 128, TPB, 0, stream>>>(v_feat, Wi, bi, v_emb, NI_, Kv);
  gemm_xwt_f16<<<(NI_ + 127) / 128, TPB, 0, stream>>>(t_feat, Wt, bt, t_emb, NI_, Kt);

  // ---- KNN-graph SpMMs + L2 norms ----
  const int spmmBlocks = (NI_ * 32 + TPB - 1) / TPB;
  spmm_knn<<<spmmBlocks, TPB, 0, stream>>>(mm_idx + 2 * nnz_img, mm_vals, item_emb,
                                           hmm, knn_k, nnz_img);
  spmm_knn<<<spmmBlocks, TPB, 0, stream>>>(img_idx + nnz_img, img_vals, v_emb,
                                           h1, knn_k, -1);
  spmm_knn<<<spmmBlocks, TPB, 0, stream>>>(txt_idx + nnz_img, txt_vals, t_emb,
                                           h2, knn_k, -1);
  l2norm_rows<<<spmmBlocks, TPB, 0, stream>>>(hmm, NI_);
  l2norm_rows<<<spmmBlocks, TPB, 0, stream>>>(h1, NI_);
  l2norm_rows<<<spmmBlocks, TPB, 0, stream>>>(h2, NI_);

  // ---- three GCN propagations ----
  // visual (filtered) -> cols 64..127
  run_propagate(user_emb, v_emb, conv_W, conv_b, eu, ev, E, 1, h1, 64,
                ego, hlin, agg, accum, deg, dinv, mask, out, stream);
  // textual (filtered) -> cols 128..191
  run_propagate(user_emb, t_emb, conv_W, conv_b, eu, ev, E, 1, h2, 128,
                ego, hlin, agg, accum, deg, dinv, mask, out, stream);
  // graph (unfiltered) -> cols 0..63
  run_propagate(user_emb, item_emb, conv_W, conv_b, eu, ev, E, 0, hmm, 0,
                ego, hlin, agg, accum, deg, dinv, mask, out, stream);
}
